// ParticleGNNImproved_25812753449156
// MI455X (gfx1250) — matrix operand order repaired and checked by reading, not compiled, not run on hardware
//
#include <hip/hip_runtime.h>
#include <math.h>

// ---------------------------------------------------------------------------
// GAT GNN forward for MI455X (gfx1250, wave32).
// Dense layers -> v_wmma_f32_16x16x32_bf16, operands pre-packed to bf16
// (activations row-major [M,Kpad], weights transposed [Nc,Kpad]) so the GEMM
// inner loop is pure global_load_b128 + v_wmma, no guards, no conversions.
// Sparse GAT attention -> L2-resident float atomics (192MB L2 holds state).
// Workspace (fp32 elems), ~50.7M floats (~203 MB):
//   ctrl[64] | X[N*128] | ID[N*128] | H[N*512] | Y[N*128]
//   | ASRC/ADST/AMAX/ASUM[N*4 each] | EX[(E+N)*4] | XB[N*64] | WT[32768]
// ---------------------------------------------------------------------------

#define NNODE 50000
#define NEDGE 400000
#define ETOT  (NEDGE + NNODE)

typedef __attribute__((ext_vector_type(16))) __bf16 v16bf;
typedef __attribute__((ext_vector_type(8)))  __bf16 v8bf;
typedef __attribute__((ext_vector_type(8)))  float  v8f;

static constexpr long long OFF_CTRL = 0;                       // 64 floats
static constexpr long long OFF_X    = 64;
static constexpr long long OFF_ID   = OFF_X  + (long long)NNODE * 128;
static constexpr long long OFF_H    = OFF_ID + (long long)NNODE * 128;
static constexpr long long OFF_Y    = OFF_H  + (long long)NNODE * 512;
static constexpr long long OFF_AS   = OFF_Y  + (long long)NNODE * 128;
static constexpr long long OFF_AD   = OFF_AS + (long long)NNODE * 4;
static constexpr long long OFF_AMAX = OFF_AD + (long long)NNODE * 4;
static constexpr long long OFF_ASUM = OFF_AMAX + (long long)NNODE * 4;
static constexpr long long OFF_EX   = OFF_ASUM + (long long)NNODE * 4;  // (E+N)*4
static constexpr long long OFF_XB   = OFF_EX + (long long)ETOT * 4;     // N*128 bf16
static constexpr long long OFF_WT   = OFF_XB + (long long)NNODE * 64;   // 65536 bf16

// ---- helpers --------------------------------------------------------------

__device__ __forceinline__ __bf16 f2bf(float f) {
  unsigned int u = __builtin_bit_cast(unsigned int, f);
  u += 0x7FFFu + ((u >> 16) & 1u);                 // round-to-nearest-even
  unsigned short s = (unsigned short)(u >> 16);
  return __builtin_bit_cast(__bf16, s);
}

__device__ __forceinline__ float wave_sum(float v) {
  #pragma unroll
  for (int off = 16; off; off >>= 1) v += __shfl_xor(v, off, 32);
  return v;
}

__device__ __forceinline__ unsigned int fenc(float x) {
  unsigned int u = __builtin_bit_cast(unsigned int, x);
  return (u & 0x80000000u) ? ~u : (u | 0x80000000u);
}
__device__ __forceinline__ float fdec(unsigned int u) {
  unsigned int b = (u & 0x80000000u) ? (u & 0x7FFFFFFFu) : ~u;
  return __builtin_bit_cast(float, b);
}

__device__ __forceinline__ void atomic_fadd(float* p, float v) {
  __hip_atomic_fetch_add(p, v, __ATOMIC_RELAXED, __HIP_MEMORY_SCOPE_AGENT);
}

__device__ __forceinline__ v16bf ld_frag(const __bf16* p0, const __bf16* p1) {
  v8bf lo = *(const v8bf*)p0;
  v8bf hi = *(const v8bf*)p1;
  return __builtin_shufflevector(lo, hi, 0, 1, 2, 3, 4, 5, 6, 7,
                                 8, 9, 10, 11, 12, 13, 14, 15);
}

// ---- operand packing ------------------------------------------------------
// activations: fp32 [M,Kin] -> bf16 [M,Kpad] row-major (zero pad)
__global__ void pack_act_kernel(const float* __restrict__ src, __bf16* __restrict__ dst,
                                int Kin, int Kpad) {
  int k = threadIdx.x;
  if (k >= Kpad) return;
  long long r = blockIdx.x;
  dst[r * Kpad + k] = f2bf(k < Kin ? src[r * Kin + k] : 0.f);
}

// weights: fp32 [Kin,Nc] -> bf16 transposed [Nc,Kpad] (zero pad)
__global__ void pack_wt_kernel(const float* __restrict__ W, __bf16* __restrict__ Wt,
                               int Kin, int Kpad, int Nc) {
  int k = threadIdx.x;
  if (k >= Kpad) return;
  long long col = blockIdx.x;
  Wt[col * Kpad + k] = f2bf(k < Kin ? W[(long long)k * Nc + col] : 0.f);
}

// ---- dense GEMM via WMMA bf16 ---------------------------------------------
// out[M,Nc] = act(A[M,K]bf16 @ Wt[Nc,K]bf16^T + bias). 16x64 strip per wave.
// K % 32 == 0, M % 16 == 0, Nc % 64 == 0. Inner loop: 10 b128 loads + 4 wmma.
template <int ACT>
__global__ __launch_bounds__(256)
void gemm_wmma(const __bf16* __restrict__ A, const __bf16* __restrict__ Bt,
               const float* __restrict__ bias, float* __restrict__ out,
               int M, int K, int Nc) {
  const int lane = threadIdx.x & 31;
  const int wave = threadIdx.x >> 5;
  const int ntn  = Nc >> 6;
  const long long tile  = (long long)blockIdx.x * 8 + wave;
  const long long total = (long long)(M >> 4) * ntn;
  if (tile >= total) return;                        // uniform per wave
  const int tm = (int)(tile / ntn);
  const int tn = (int)(tile % ntn);

  // A fragment (16-bit A 16x32, contiguous k order): lane row = tm*16+(lane&15),
  // K offsets {khalfA+0..15}, khalfA = 16*(lane>=16)
  const int khalfA = (lane >> 4) << 4;
  const int khalfB = (lane >> 4) << 4;              // B: K = e + 16*(lane>=16)
  const __bf16* ap = A + (long long)(tm * 16 + (lane & 15)) * K + khalfA;
  const int col0 = tn * 64 + (lane & 15);
  const __bf16* bp0 = Bt + (long long)(col0)      * K + khalfB;
  const __bf16* bp1 = Bt + (long long)(col0 + 16) * K + khalfB;
  const __bf16* bp2 = Bt + (long long)(col0 + 32) * K + khalfB;
  const __bf16* bp3 = Bt + (long long)(col0 + 48) * K + khalfB;

  v8f acc0 = {}, acc1 = {}, acc2 = {}, acc3 = {};
  for (int k0 = 0; k0 < K; k0 += 32) {
    v16bf af = ld_frag(ap + k0, ap + k0 + 8);
    v16bf b0 = ld_frag(bp0 + k0, bp0 + k0 + 8);
    v16bf b1 = ld_frag(bp1 + k0, bp1 + k0 + 8);
    v16bf b2 = ld_frag(bp2 + k0, bp2 + k0 + 8);
    v16bf b3 = ld_frag(bp3 + k0, bp3 + k0 + 8);
    acc0 = __builtin_amdgcn_wmma_f32_16x16x32_bf16(false, af, false, b0, (short)0, acc0, false, false);
    acc1 = __builtin_amdgcn_wmma_f32_16x16x32_bf16(false, af, false, b1, (short)0, acc1, false, false);
    acc2 = __builtin_amdgcn_wmma_f32_16x16x32_bf16(false, af, false, b2, (short)0, acc2, false, false);
    acc3 = __builtin_amdgcn_wmma_f32_16x16x32_bf16(false, af, false, b3, (short)0, acc3, false, false);
  }

  // D layout: n = lane&15, m = vgpr + 8*(lane>=16)
  const int orow = tm * 16 + ((lane >> 4) << 3);
  #pragma unroll
  for (int j = 0; j < 4; ++j) {
    const v8f& a = (j == 0) ? acc0 : (j == 1) ? acc1 : (j == 2) ? acc2 : acc3;
    const int col = tn * 64 + j * 16 + (lane & 15);
    const float bv = bias ? bias[col] : 0.f;
    #pragma unroll
    for (int v = 0; v < 8; ++v) {
      float r = a[v] + bv;
      if (ACT == 1) r = fmaxf(r, 0.f);
      out[(long long)(orow + v) * Nc + col] = r;
    }
  }
}

// ---- edge_attr mean -------------------------------------------------------
__global__ void ea_sum_kernel(const float* __restrict__ ea, float* __restrict__ ctrl) {
  int i = blockIdx.x * blockDim.x + threadIdx.x;
  float v = (i < NEDGE) ? ea[i] : 0.f;
  v = wave_sum(v);
  if ((threadIdx.x & 31) == 0) atomic_fadd(&ctrl[0], v);
}

// ---- we_ae[h] = dot(We[h*C:], ae[h*C:]) -----------------------------------
__global__ void we_ae_kernel(const float* __restrict__ We, const float* __restrict__ ae,
                             float* __restrict__ out, int C) {
  int h = threadIdx.x >> 5, lane = threadIdx.x & 31;
  float s = 0.f;
  for (int c = lane; c < C; c += 32) s += We[h * C + c] * ae[h * C + c];
  s = wave_sum(s);
  if (lane == 0) out[h] = s;
}

// ---- per-node attention pre-dots ------------------------------------------
__global__ void node_att_kernel(const float* __restrict__ hbuf,
                                const float* __restrict__ as_, const float* __restrict__ ad_,
                                float* __restrict__ asrc, float* __restrict__ adst, int C) {
  long long idx = (long long)blockIdx.x * blockDim.x + threadIdx.x;
  if (idx >= (long long)NNODE * 4) return;
  int n = (int)(idx >> 2), h = (int)(idx & 3);
  const float* hp = hbuf + ((long long)n * 4 + h) * C;
  float s1 = 0.f, s2 = 0.f;
  for (int c = 0; c < C; ++c) { float hv = hp[c]; s1 += hv * as_[h * C + c]; s2 += hv * ad_[h * C + c]; }
  asrc[idx] = s1; adst[idx] = s2;
}

// ---- pass 1: logits + segment max -----------------------------------------
__global__ void edge_logit_max(const int* __restrict__ ei, const float* __restrict__ ea,
                               const float* __restrict__ ctrl,
                               const float* __restrict__ asrc, const float* __restrict__ adst,
                               const float* __restrict__ we_ae,
                               float* __restrict__ exbuf, unsigned int* __restrict__ amax) {
  long long idx = (long long)blockIdx.x * blockDim.x + threadIdx.x;
  if (idx >= (long long)ETOT * 4) return;
  int e = (int)(idx >> 2), h = (int)(idx & 3);
  int s, d; float eav;
  if (e < NEDGE) { s = ei[e]; d = ei[NEDGE + e]; eav = ea[e]; }
  else           { s = d = e - NEDGE; eav = ctrl[0] * (1.0f / NEDGE); }
  float a = asrc[(long long)s * 4 + h] + adst[(long long)d * 4 + h] + eav * we_ae[h];
  a = (a > 0.f) ? a : 0.2f * a;                    // leaky_relu
  exbuf[idx] = a;
  atomicMax(&amax[(long long)d * 4 + h], fenc(a));
}

// ---- pass 2: exp + segment sum --------------------------------------------
__global__ void edge_expsum(const int* __restrict__ ei,
                            float* __restrict__ exbuf, const unsigned int* __restrict__ amax,
                            float* __restrict__ asum) {
  long long idx = (long long)blockIdx.x * blockDim.x + threadIdx.x;
  if (idx >= (long long)ETOT * 4) return;
  int e = (int)(idx >> 2), h = (int)(idx & 3);
  int d = (e < NEDGE) ? ei[NEDGE + e] : (e - NEDGE);
  float m  = fdec(amax[(long long)d * 4 + h]);
  float ex = __expf(exbuf[idx] - m);
  exbuf[idx] = ex;
  atomic_fadd(&asum[(long long)d * 4 + h], ex);
}

// ---- pass 3: weighted aggregation (head-mean fused, one wave per edge) -----
__global__ __launch_bounds__(256)
void edge_aggregate(const int* __restrict__ ei, const float* __restrict__ exbuf,
                    const float* __restrict__ asum, const float* __restrict__ hbuf,
                    float* __restrict__ ymean, int C) {
  long long wid = ((long long)blockIdx.x * blockDim.x + threadIdx.x) >> 5;
  int lane = threadIdx.x & 31;
  if (wid >= ETOT) return;                         // uniform per wave
  int e = (int)wid;
  int s, d;
  if (e < NEDGE) { s = ei[e]; d = ei[NEDGE + e]; } else { s = d = e - NEDGE; }
  float al[4];
  #pragma unroll
  for (int h = 0; h < 4; ++h)
    al[h] = 0.25f * exbuf[(long long)e * 4 + h] /
            (asum[(long long)d * 4 + h] + 1e-16f);
  const float* hp = hbuf + (long long)s * 4 * C;
  float* yp = ymean + (long long)d * C;
  for (int c = lane; c < C; c += 32) {
    float v = al[0] * hp[c] + al[1] * hp[C + c] + al[2] * hp[2 * C + c] + al[3] * hp[3 * C + c];
    atomic_fadd(&yp[c], v);
  }
}

// ---- epilogue: (+bias) -> LayerNorm -> (+identity) -> ReLU ----------------
__global__ __launch_bounds__(256)
void node_epilogue(const float* __restrict__ src, const float* __restrict__ bias,
                   const float* __restrict__ g, const float* __restrict__ b,
                   const float* __restrict__ ident, float* __restrict__ dst, int C) {
  long long node = ((long long)blockIdx.x * (blockDim.x >> 5)) + (threadIdx.x >> 5);
  int lane = threadIdx.x & 31;
  if (node >= NNODE) return;                       // uniform per wave
  const int nc = C >> 5;                           // 2 or 4 chans per lane
  float vals[4];
  #pragma unroll 4
  for (int j = 0; j < nc; ++j) {
    int c = lane + (j << 5);
    vals[j] = src[node * C + c] + (bias ? bias[c] : 0.f);
  }
  float s = 0.f;
  for (int j = 0; j < nc; ++j) s += vals[j];
  float mean = wave_sum(s) * (1.0f / C);
  float vsum = 0.f;
  for (int j = 0; j < nc; ++j) { float t = vals[j] - mean; vsum += t * t; }
  float inv = rsqrtf(wave_sum(vsum) * (1.0f / C) + 1e-5f);
  #pragma unroll 4
  for (int j = 0; j < nc; ++j) {
    int c = lane + (j << 5);
    float y = (vals[j] - mean) * inv * g[c] + b[c];
    if (ident) y += ident[node * C + c];
    dst[node * C + c] = fmaxf(y, 0.f);
  }
}

// ---- classifier head: f3 + log_softmax ------------------------------------
__global__ void classifier_out(const float* __restrict__ xin, const float* __restrict__ W,
                               const float* __restrict__ b, float* __restrict__ out) {
  int n = blockIdx.x * blockDim.x + threadIdx.x;
  if (n >= NNODE) return;
  float z0 = b[0], z1 = b[1], z2 = b[2];
  for (int c = 0; c < 64; ++c) {
    float xv = xin[(long long)n * 64 + c];
    z0 += xv * W[c * 3 + 0]; z1 += xv * W[c * 3 + 1]; z2 += xv * W[c * 3 + 2];
  }
  float m = fmaxf(z0, fmaxf(z1, z2));
  float l = m + __logf(__expf(z0 - m) + __expf(z1 - m) + __expf(z2 - m));
  out[(long long)n * 3 + 0] = z0 - l;
  out[(long long)n * 3 + 1] = z1 - l;
  out[(long long)n * 3 + 2] = z2 - l;
}

// ---- host orchestration ---------------------------------------------------

static void launch_gemm(hipStream_t s, const __bf16* A, const __bf16* Wt, const float* bias,
                        float* out, int M, int K, int Nc, int relu) {
  long long tiles = (long long)(M / 16) * (Nc / 64);
  unsigned blocks = (unsigned)((tiles + 7) / 8);
  if (relu) gemm_wmma<1><<<blocks, 256, 0, s>>>(A, Wt, bias, out, M, K, Nc);
  else      gemm_wmma<0><<<blocks, 256, 0, s>>>(A, Wt, bias, out, M, K, Nc);
}

static void pack_act(hipStream_t s, const float* src, __bf16* dst, int M, int Kin, int Kpad) {
  pack_act_kernel<<<M, 128, 0, s>>>(src, dst, Kin, Kpad);
}
static void pack_wt(hipStream_t s, const float* W, __bf16* Wt, int Kin, int Kpad, int Nc) {
  pack_wt_kernel<<<Nc, 128, 0, s>>>(W, Wt, Kin, Kpad, Nc);
}

static void run_conv(hipStream_t st, float* ws, const int* ei, const float* ea,
                     const __bf16* xb, int Kpad, int C,
                     const float* W, const float* We, const float* as_, const float* ad_,
                     const float* ae_, const float* cb, const float* g, const float* b,
                     const float* ident, float* xout) {
  float* ctrl  = ws + OFF_CTRL;
  float* hbuf  = ws + OFF_H;
  float* ymean = ws + OFF_Y;
  float* asrc  = ws + OFF_AS;
  float* adst  = ws + OFF_AD;
  unsigned int* amax = (unsigned int*)(ws + OFF_AMAX);
  float* asum  = ws + OFF_ASUM;
  float* exbuf = ws + OFF_EX;
  __bf16* WT   = (__bf16*)(ws + OFF_WT);

  pack_wt(st, W, WT, Kpad, Kpad, 4 * C);                               // fin == Kpad here
  launch_gemm(st, xb, WT, nullptr, hbuf, NNODE, Kpad, 4 * C, 0);       // h = x @ W
  we_ae_kernel<<<1, 128, 0, st>>>(We, ae_, ctrl + 8, C);
  hipMemsetAsync(ymean, 0, (size_t)NNODE * C * sizeof(float), st);
  hipMemsetAsync(ws + OFF_AMAX, 0, (size_t)NNODE * 8 * sizeof(float), st);
  node_att_kernel<<<(NNODE * 4 + 255) / 256, 256, 0, st>>>(hbuf, as_, ad_, asrc, adst, C);

  unsigned egrid = (unsigned)(((long long)ETOT * 4 + 255) / 256);
  edge_logit_max<<<egrid, 256, 0, st>>>(ei, ea, ctrl, asrc, adst, ctrl + 8, exbuf, amax);
  edge_expsum<<<egrid, 256, 0, st>>>(ei, exbuf, amax, asum);
  edge_aggregate<<<(ETOT + 7) / 8, 256, 0, st>>>(ei, exbuf, asum, hbuf, ymean, C);
  node_epilogue<<<(NNODE + 7) / 8, 256, 0, st>>>(ymean, cb, g, b, ident, xout, C);
}

extern "C" void kernel_launch(void* const* d_in, const int* in_sizes, int n_in,
                              void* d_out, int out_size, void* d_ws, size_t ws_size,
                              hipStream_t stream) {
  (void)in_sizes; (void)n_in; (void)out_size; (void)ws_size;
  auto F = [&](int i) { return (const float*)d_in[i]; };
  const float* x  = F(0);
  const int*   ei = (const int*)d_in[1];
  const float* ea = F(2);
  const float *in_W = F(3), *in_b = F(4);
  const float *c1W = F(5), *c1We = F(6), *c1as = F(7), *c1ad = F(8), *c1ae = F(9), *c1b = F(10);
  const float *n1g = F(11), *n1b = F(12);
  const float *p2W = F(13), *p2b = F(14);
  const float *c2W = F(15), *c2We = F(16), *c2as = F(17), *c2ad = F(18), *c2ae = F(19), *c2b = F(20);
  const float *n2g = F(21), *n2b = F(22);
  const float *c3W = F(23), *c3We = F(24), *c3as = F(25), *c3ad = F(26), *c3ae = F(27), *c3b = F(28);
  const float *n3g = F(29), *n3b = F(30);
  const float *p4W = F(31), *p4b = F(32);
  const float *c4W = F(33), *c4We = F(34), *c4as = F(35), *c4ad = F(36), *c4ae = F(37), *c4b = F(38);
  const float *n4g = F(39), *n4b = F(40);
  const float *f1W = F(41), *f1b = F(42), *fn1g = F(43), *fn1b = F(44);
  const float *f2W = F(45), *f2b = F(46), *fn2g = F(47), *fn2b = F(48);
  const float *f3W = F(49), *f3b = F(50);

  float* ws = (float*)d_ws;
  float* X  = ws + OFF_X;
  float* ID = ws + OFF_ID;
  float* H  = ws + OFF_H;
  float* Y  = ws + OFF_Y;
  __bf16* XB = (__bf16*)(ws + OFF_XB);
  __bf16* WT = (__bf16*)(ws + OFF_WT);

  hipMemsetAsync(ws, 0, 64 * sizeof(float), stream);
  ea_sum_kernel<<<(NEDGE + 255) / 256, 256, 0, stream>>>(ea, ws);

  // input projection: X = relu(x @ in_W + in_b)   [N,16]->[N,64], K padded to 32
  pack_act(stream, x, XB, NNODE, 16, 32);
  pack_wt(stream, in_W, WT, 16, 32, 64);
  launch_gemm(stream, XB, WT, in_b, X, NNODE, 32, 64, 1);

  // conv1: fin=64, C=64, identity = X
  pack_act(stream, X, XB, NNODE, 64, 64);
  run_conv(stream, ws, ei, ea, XB, 64, 64, c1W, c1We, c1as, c1ad, c1ae, c1b, n1g, n1b, X, X);

  // conv2: identity = X @ p2_W + p2_b (64->128); conv fin=64, C=128
  pack_act(stream, X, XB, NNODE, 64, 64);
  pack_wt(stream, p2W, WT, 64, 64, 128);
  launch_gemm(stream, XB, WT, p2b, ID, NNODE, 64, 128, 0);
  run_conv(stream, ws, ei, ea, XB, 64, 128, c2W, c2We, c2as, c2ad, c2ae, c2b, n2g, n2b, ID, X);

  // conv3: fin=128, C=128, identity = X
  pack_act(stream, X, XB, NNODE, 128, 128);
  run_conv(stream, ws, ei, ea, XB, 128, 128, c3W, c3We, c3as, c3ad, c3ae, c3b, n3g, n3b, X, X);

  // conv4: identity = X @ p4_W + p4_b (128->64); conv fin=128, C=64
  pack_act(stream, X, XB, NNODE, 128, 128);
  pack_wt(stream, p4W, WT, 128, 128, 64);
  launch_gemm(stream, XB, WT, p4b, ID, NNODE, 128, 64, 0);
  run_conv(stream, ws, ei, ea, XB, 128, 64, c4W, c4We, c4as, c4ad, c4ae, c4b, n4g, n4b, ID, X);

  // classifier MLP
  pack_act(stream, X, XB, NNODE, 64, 64);
  pack_wt(stream, f1W, WT, 64, 64, 128);
  launch_gemm(stream, XB, WT, f1b, H, NNODE, 64, 128, 0);
  node_epilogue<<<(NNODE + 7) / 8, 256, 0, stream>>>(H, nullptr, fn1g, fn1b, nullptr, H, 128);
  pack_act(stream, H, XB, NNODE, 128, 128);
  pack_wt(stream, f2W, WT, 128, 128, 64);
  launch_gemm(stream, XB, WT, f2b, Y, NNODE, 128, 64, 0);
  node_epilogue<<<(NNODE + 7) / 8, 256, 0, stream>>>(Y, nullptr, fn2g, fn2b, nullptr, Y, 64);
  classifier_out<<<(NNODE + 255) / 256, 256, 0, stream>>>(Y, f3W, f3b, (float*)d_out);
}
